// Chn8ActGrp3WgtQuantizedLinear_54485955117162
// MI455X (gfx1250) — 8-bit matrix operand order repaired and checked by reading, compiled, not run on hardware
//
#include <hip/hip_runtime.h>
#include <stdint.h>

#define MDIM 1024
#define KDIM 4096
#define NDIM 11008
#define NGRP 32
#define BM   128
#define BN   128
#define CHK  64
#define ROWB 80

typedef int v8i __attribute__((ext_vector_type(8)));

#if defined(__has_builtin)
#  if __has_builtin(__builtin_amdgcn_sched_barrier)
#    define SCHED_FENCE() __builtin_amdgcn_sched_barrier(0)
#  endif
#endif
#ifndef SCHED_FENCE
#  define SCHED_FENCE() asm volatile("" ::: "memory")
#endif

#if defined(__has_builtin) && __has_builtin(__builtin_amdgcn_s_wait_asynccnt)
#  define ASYNC_WAIT0() __builtin_amdgcn_s_wait_asynccnt(0)
#else
#  define ASYNC_WAIT0() asm volatile("s_wait_asynccnt 0x0" ::: "memory")
#endif

__device__ __forceinline__ void async_copy32(unsigned lds_addr, const void* gptr) {
  asm volatile(
      "global_load_async_to_lds_b128 %0, %1, off\n\t"
      "global_load_async_to_lds_b128 %0, %1, off offset:16"
      :: "v"(lds_addr), "v"(gptr) : "memory");
}

__device__ __forceinline__ int pack4(int4 w) {
#if defined(__has_builtin) && __has_builtin(__builtin_amdgcn_perm)
  unsigned t0 = __builtin_amdgcn_perm((unsigned)w.y, (unsigned)w.x, 0x00000400u);
  unsigned t1 = __builtin_amdgcn_perm((unsigned)w.w, (unsigned)w.z, 0x00000400u);
  return (int)__builtin_amdgcn_perm(t1, t0, 0x05040100u);
#else
  return (w.x & 255) | ((w.y & 255) << 8) | ((w.z & 255) << 16) | (w.w << 24);
#endif
}

__global__ __launch_bounds__(256) void quant_act(const float* __restrict__ x,
                                                 int8_t* __restrict__ q,
                                                 float* __restrict__ sx,
                                                 int* __restrict__ zp) {
  __shared__ float rmn[256], rmx[256];
  const int m = blockIdx.x, t = threadIdx.x;
  const float4* xr = (const float4*)(x + (size_t)m * KDIM);

  float vals[16];
  float vmin = 0.0f, vmax = 0.0f;
#pragma unroll
  for (int j = 0; j < 4; ++j) {
    float4 v = xr[t * 4 + j];
    vals[j * 4 + 0] = v.x; vals[j * 4 + 1] = v.y;
    vals[j * 4 + 2] = v.z; vals[j * 4 + 3] = v.w;
    vmin = fminf(vmin, fminf(fminf(v.x, v.y), fminf(v.z, v.w)));
    vmax = fmaxf(vmax, fmaxf(fmaxf(v.x, v.y), fmaxf(v.z, v.w)));
  }
  rmn[t] = vmin; rmx[t] = vmax;
  __syncthreads();
  for (int s = 128; s > 0; s >>= 1) {
    if (t < s) {
      rmn[t] = fminf(rmn[t], rmn[t + s]);
      rmx[t] = fmaxf(rmx[t], rmx[t + s]);
    }
    __syncthreads();
  }
  const float xmin = rmn[0], xmax = rmx[0];
  const float scale = fmaxf((xmax - xmin) / 255.0f, 1e-9f);
  const float zf = rintf(-128.0f - xmin / scale);
  if (t == 0) { sx[m] = scale; zp[m] = (int)zf; }

  int pk[4];
#pragma unroll
  for (int j = 0; j < 4; ++j) {
    int b[4];
#pragma unroll
    for (int e = 0; e < 4; ++e) {
      float qv = rintf(vals[j * 4 + e] / scale) + zf;
      qv = fminf(fmaxf(qv, -128.0f), 127.0f);
      b[e] = (int)qv;
    }
    pk[j] = (b[0] & 255) | ((b[1] & 255) << 8) | ((b[2] & 255) << 16) | (b[3] << 24);
  }
  ((int4*)(q + (size_t)m * KDIM))[t] = make_int4(pk[0], pk[1], pk[2], pk[3]);
}

__global__ __launch_bounds__(256) void gemm_w3a8(
    const int8_t* __restrict__ q, const float* __restrict__ sx,
    const int* __restrict__ zp, const int* __restrict__ wq,
    const float* __restrict__ wsc, float* __restrict__ out) {
  __shared__ int8_t Ald[2][BM * ROWB];
  __shared__ int8_t Bld[2][BN * ROWB];
  __shared__ int    csp[2][2 * BN];
  __shared__ float  sx_l[BM];
  __shared__ int    zp_l[BM];

  const int t    = threadIdx.x;
  const int lane = t & 31;
  const int wave = t >> 5;
  const int ln   = lane & 15;
  const int hi   = lane >> 4;
  const int wm   = wave >> 1;
  const int wn   = wave & 1;
  const int mblk = blockIdx.x * BM;
  const int nblk = blockIdx.y * BN;

  if (t < BM) { sx_l[t] = sx[mblk + t]; zp_l[t] = zp[mblk + t]; }
  __syncthreads();

  float sxv[2][8];
#pragma unroll
  for (int mt = 0; mt < 2; ++mt)
#pragma unroll
    for (int r = 0; r < 8; ++r)
      sxv[mt][r] = sx_l[wm * 32 + mt * 16 + r + hi * 8];

  v8i   iacc[2][4];
  float facc[2][4][8];
  int   csacc[4];
  float cwsum[4];
#pragma unroll
  for (int mt = 0; mt < 2; ++mt)
#pragma unroll
    for (int nt = 0; nt < 4; ++nt)
#pragma unroll
      for (int r = 0; r < 8; ++r) { iacc[mt][nt][r] = 0; facc[mt][nt][r] = 0.0f; }
#pragma unroll
  for (int nt = 0; nt < 4; ++nt) { csacc[nt] = 0; cwsum[nt] = 0.0f; }

  const int rl = t & 127;
  const int hf = t >> 7;
  const int8_t* gA = q  + (size_t)(mblk + rl) * KDIM + hf * 32;
  const int*    gB = wq + (size_t)(nblk + rl) * KDIM + hf * 32;
  int8_t* sB[2] = { Bld[0] + rl * ROWB + hf * 32, Bld[1] + rl * ROWB + hf * 32 };
  int*    sC[2] = { &csp[0][hf * BN + rl], &csp[1][hf * BN + rl] };
  const unsigned aLds[2] = {
    (unsigned)(uintptr_t)(Ald[0] + rl * ROWB + hf * 32),
    (unsigned)(uintptr_t)(Ald[1] + rl * ROWB + hf * 32)
  };

  const float* gS = wsc + (size_t)(nblk + wn * 64 + ln) * NGRP;

  int4  pf[8];
  float swv[4];

  auto prefetch = [&](int kc, int buf) {
    const int ko = kc * CHK;
    async_copy32(aLds[buf], gA + ko);
#pragma unroll
    for (int j = 0; j < 8; ++j) pf[j] = *(const int4*)(gB + ko + j * 4);
  };

  auto stage = [&](int buf) {
    int pk[8], cs = 0;
#pragma unroll
    for (int j = 0; j < 8; ++j) {
      cs += pf[j].x + pf[j].y + pf[j].z + pf[j].w;
      pk[j] = pack4(pf[j]);
    }
    *(int4*)sB[buf]        = make_int4(pk[0], pk[1], pk[2], pk[3]);
    *(int4*)(sB[buf] + 16) = make_int4(pk[4], pk[5], pk[6], pk[7]);
    *sC[buf] = cs;
  };

  auto compute = [&](int buf) {
    const int8_t* Ab  = Ald[buf];
    const int8_t* Bb  = Bld[buf];
    const int*    csb = csp[buf];
    v8i af[2];
#pragma unroll
    for (int mt = 0; mt < 2; ++mt) {
      const int8_t* ar = Ab + (wm * 32 + mt * 16 + ln) * ROWB + hi * 16;
#pragma unroll
      for (int p = 0; p < 4; ++p) {
        const int2 d = *(const int2*)(ar + p * 8 + (p >> 1) * 16);
        af[mt][2 * p]     = d.x;
        af[mt][2 * p + 1] = d.y;
      }
    }
#pragma unroll
    for (int nt = 0; nt < 4; ++nt) {
      const int nl = wn * 64 + nt * 16 + ln;
      const int8_t* br = Bb + nl * ROWB + hi * 16;
      const int4 b0 = *(const int4*)br;
      const int4 b1 = *(const int4*)(br + 32);
      v8i bf;
      bf[0] = b0.x; bf[1] = b0.y; bf[2] = b0.z; bf[3] = b0.w;
      bf[4] = b1.x; bf[5] = b1.y; bf[6] = b1.z; bf[7] = b1.w;
      csacc[nt] += csb[nl] + csb[BN + nl];
#pragma unroll
      for (int mt = 0; mt < 2; ++mt)
        iacc[mt][nt] = __builtin_amdgcn_wmma_i32_16x16x64_iu8(
             true, af[mt],  true, bf, iacc[mt][nt],
             false,  false);
    }
  };

  prefetch(0, 0);
  stage(0);

  for (int g = 0; g < NGRP; ++g) {
    ASYNC_WAIT0();
    __syncthreads();
    prefetch(2 * g + 1, 1);
    SCHED_FENCE();
    compute(0);
    SCHED_FENCE();
    stage(1);

    ASYNC_WAIT0();
    __syncthreads();
    int knext = 2 * g + 2;
    if (knext > KDIM / CHK - 1) knext = KDIM / CHK - 1;
    prefetch(knext, 0);
#pragma unroll
    for (int nt = 0; nt < 4; ++nt) swv[nt] = gS[(size_t)(nt * 16) * NGRP + g];
    SCHED_FENCE();
    compute(1);
    SCHED_FENCE();

#pragma unroll
    for (int nt = 0; nt < 4; ++nt) {
      const float sw = swv[nt];
      cwsum[nt] = fmaf((float)csacc[nt], sw, cwsum[nt]);
      csacc[nt] = 0;
#pragma unroll
      for (int mt = 0; mt < 2; ++mt) {
        const float* s = sxv[mt];
#pragma unroll
        for (int r = 0; r < 8; ++r) {
          facc[mt][nt][r] = fmaf((float)iacc[mt][nt][r], s[r] * sw, facc[mt][nt][r]);
          iacc[mt][nt][r] = 0;
        }
      }
    }

    stage(0);
  }

  float zsxv[2][8];
#pragma unroll
  for (int mt = 0; mt < 2; ++mt)
#pragma unroll
    for (int r = 0; r < 8; ++r) {
      const int ml = wm * 32 + mt * 16 + r + hi * 8;
      zsxv[mt][r] = (float)zp_l[ml] * sx_l[ml];
    }
#pragma unroll
  for (int mt = 0; mt < 2; ++mt)
#pragma unroll
    for (int nt = 0; nt < 4; ++nt) {
      const int n = nblk + wn * 64 + nt * 16 + ln;
#pragma unroll
      for (int r = 0; r < 8; ++r) {
        const int m = mblk + wm * 32 + mt * 16 + r + hi * 8;
        const float v = fmaf(-zsxv[mt][r], cwsum[nt], facc[mt][nt][r]);
        __builtin_nontemporal_store(v, &out[(size_t)m * NDIM + n]);
      }
    }
}

extern "C" void kernel_launch(void* const* d_in, const int* in_sizes, int n_in,
                              void* d_out, int out_size, void* d_ws, size_t ws_size,
                              hipStream_t stream) {
  const float* x   = (const float*)d_in[0];
  const int*   wq  = (const int*)d_in[1];
  const float* wsc = (const float*)d_in[2];
  (void)in_sizes; (void)n_in; (void)out_size; (void)ws_size;

  int8_t* q  = (int8_t*)d_ws;
  float*  sx = (float*)((char*)d_ws + (size_t)MDIM * KDIM);
  int*    zp = (int*)((char*)sx + (size_t)MDIM * sizeof(float));
  float*  out = (float*)d_out;

  quant_act<<<MDIM, 256, 0, stream>>>(x, q, sx, zp);
  gemm_w3a8<<<dim3(MDIM / BM, NDIM / BN), 256, 0, stream>>>(q, sx, zp, wq, wsc, out);
}
